// DiscreteEncoder_43791486550204
// MI455X (gfx1250) — matrix operand order repaired and checked by reading, not compiled, not run on hardware
//
#include <hip/hip_runtime.h>
#include <hip/hip_bf16.h>

// Problem constants (match reference)
#define NN 100000
#define DD 300
#define DP 320            // D padded to multiple of 32 (bf16 WMMA K-chunks)
#define KK 4096
#define CC 10
#define GG 1000
#define COMMIT_WF 1.0f
#define KCH (DP / 32)     // 10 K-chunks per dot product
#define SLICES 16
#define KS 256            // codes per slice; B slice in LDS = 256*320*2 = 160KB
#define VQ_WAVES 4
#define CHUNKS 64
#define PAIRS (NN / 32)   // 3125 pairs of 16-row tiles (N = 3125*32 exactly)

typedef __attribute__((ext_vector_type(16))) __bf16 v16bf;
typedef __attribute__((ext_vector_type(8)))  float  v8f;
typedef __attribute__((ext_vector_type(4)))  unsigned int u32x4;
typedef __attribute__((ext_vector_type(8)))  int    i32x8;
typedef __attribute__((ext_vector_type(4)))  int    i32x4;

static __device__ __forceinline__ unsigned short f32_to_bf16_rne(float f) {
  unsigned int u = __float_as_uint(f);
  unsigned int r = (u + 0x7FFFu + ((u >> 16) & 1u)) >> 16;
  return (unsigned short)r;
}

// ---------------- Kernel 1: codebook fp32 -> bf16, padded to DP ----------------
__global__ __launch_bounds__(256) void cb_convert_kernel(
    const float* __restrict__ cb, unsigned short* __restrict__ cbh) {
  int i = blockIdx.x * 256 + threadIdx.x;           // over K*DP
  if (i >= KK * DP) return;
  int k = i / DP, d = i % DP;
  float v = (d < DD) ? cb[(size_t)k * DD + d] : 0.0f;
  cbh[i] = f32_to_bf16_rne(v);
}

// ---------------- Kernel 2: half squared norms of codebook rows ----------------
__global__ __launch_bounds__(256) void cb_norms_kernel(
    const float* __restrict__ cb, float* __restrict__ hn) {
  int wave = threadIdx.x >> 5, lane = threadIdx.x & 31;
  int k = blockIdx.x * 8 + wave;
  if (k >= KK) return;
  float s = 0.0f;
  for (int d = lane; d < DD; d += 32) {
    float v = cb[(size_t)k * DD + d];
    s += v * v;
  }
  #pragma unroll
  for (int m = 16; m >= 1; m >>= 1) s += __shfl_xor(s, m, 32);
  if (lane == 0) hn[k] = 0.5f * s;
}

// ---------------- Kernel 3: per-graph start offsets (batch sorted) -------------
__global__ __launch_bounds__(256) void seg_starts_kernel(
    const int* __restrict__ batch, int* __restrict__ starts) {
  int g = blockIdx.x * 256 + threadIdx.x;
  if (g > GG) return;
  if (g == GG) { starts[GG] = NN; return; }
  int lo = 0, hi = NN;
  while (lo < hi) { int mid = (lo + hi) >> 1; if (batch[mid] < g) lo = mid + 1; else hi = mid; }
  starts[g] = lo;
}

// ---------------- Kernel 4: codebook-stationary VQ argmax (bf16 WMMA) ----------
// grid = (CHUNKS, SLICES). Each block TDM-loads a 256-code bf16 slice into LDS
// once, then its 4 waves sweep node-tile pairs (32 rows each, A in registers).
__global__ __launch_bounds__(VQ_WAVES * 32) void vq_argmax_kernel(
    const float* __restrict__ nf, const unsigned short* __restrict__ cbh,
    const float* __restrict__ hn, float* __restrict__ pv, int* __restrict__ pi) {
  extern __shared__ unsigned short smem[];   // [KS*DP] B slice | per-wave A staging
  const int wave  = threadIdx.x >> 5;
  const int lane  = threadIdx.x & 31;
  const int slice = blockIdx.y;
  const int chunk = blockIdx.x;

  // ---- Tensor Data Mover: DMA codebook slice (2D tile 256 x 320 of 2B) to LDS.
  if (threadIdx.x < 32) {                    // wave 0 issues the TDM op
    unsigned long long ga =
        (unsigned long long)(const void*)(cbh + (size_t)slice * KS * DP);
    // D# group0: count=1 | lds_addr=0 | global_addr | type=2
    u32x4 g0 = { 1u, 0u, (unsigned)ga,
                 (unsigned)((ga >> 32) & 0x01FFFFFFu) | 0x80000000u };
    // D# group1: data_size=2B | tensor_dim0=320 | tensor_dim1=4096 |
    //            tile_dim0=320 | tile_dim1=256 | tensor_dim0_stride=320
    i32x8 g1 = { (int)(1u << 16),        // data_size = 1 (2 bytes)
                 (int)(0x140u << 16),    // tensor_dim0[15:0] = 320
                 (int)(0x1000u << 16),   // tensor_dim1[15:0] = 4096
                 (int)(0x140u << 16),    // tile_dim0 = 320
                 (int)KS,                // tile_dim1 = 256
                 (int)DP,                // tensor_dim0_stride = 320
                 0, 0 };
    i32x4 gz4 = { 0, 0, 0, 0 };
    i32x8 gz8 = { 0, 0, 0, 0, 0, 0, 0, 0 };
    __builtin_amdgcn_tensor_load_to_lds(g0, g1, gz4, gz4, gz8, 0);
    __builtin_amdgcn_s_wait_tensorcnt(0);
  }
  __syncthreads();

  unsigned short* Abase = smem + KS * DP + wave * (2 * 16 * DP);

  // Wave32 WMMA fragment addressing (contiguous k = 16h + s, same for A and B):
  const int rowA  = lane & 15;
  const int kbA   = (lane < 16) ? 0 : 16;
  const int kbB   = (lane < 16) ? 0 : 16;
  const int colLn = lane & 15;
  const int codeBase = slice * KS + colLn;

  for (int p = chunk * VQ_WAVES + wave; p < PAIRS; p += CHUNKS * VQ_WAVES) {
    const int row0 = p * 32;
    // Stage A pair: 32 rows x DP, fp32 -> bf16, zero-padded (once per pair)
    for (int r = 0; r < 32; ++r) {
      const float* src = nf + (size_t)(row0 + r) * DD;
      for (int d = lane; d < DP; d += 32)
        Abase[r * DP + d] = f32_to_bf16_rne(d < DD ? src[d] : 0.0f);
    }
    __builtin_amdgcn_wave_barrier();   // per-wave LDS RAW; DS is in-order per wave

    // A fragments for both tiles -> registers (reused across all 160 WMMAs)
    v16bf a0[KCH], a1[KCH];
    #pragma unroll
    for (int kc = 0; kc < KCH; ++kc) {
      union { v16bf v; uint4 q[2]; } f;
      const unsigned short* ap = Abase + rowA * DP + kc * 32 + kbA;
      f.q[0] = *(const uint4*)ap;        f.q[1] = *(const uint4*)(ap + 8);
      a0[kc] = f.v;
      ap += 16 * DP;
      f.q[0] = *(const uint4*)ap;        f.q[1] = *(const uint4*)(ap + 8);
      a1[kc] = f.v;
    }

    float bv0[8], bv1[8];
    int   bi0[8], bi1[8];
    #pragma unroll
    for (int i = 0; i < 8; ++i) {
      bv0[i] = -3.0e38f; bv1[i] = -3.0e38f; bi0[i] = 0; bi1[i] = 0;
    }

    for (int ct = 0; ct < KS / 16; ++ct) {
      const int code = codeBase + ct * 16;
      const unsigned short* brow = smem + (ct * 16 + colLn) * DP;
      const float h = hn[code];          // issue early: hidden behind 20 WMMAs

      // Software-pipelined B fragments: depth-2 DS prefetch, 3 rotating regs.
      union Frag { v16bf v; uint4 q[2]; };
      Frag fb[3];
      {
        const uint4* bp0 = (const uint4*)(brow + 0 * 32 + kbB);
        fb[0].q[0] = bp0[0]; fb[0].q[1] = bp0[1];
        const uint4* bp1 = (const uint4*)(brow + 1 * 32 + kbB);
        fb[1].q[0] = bp1[0]; fb[1].q[1] = bp1[1];
      }
      v8f acc0 = {0.f,0.f,0.f,0.f,0.f,0.f,0.f,0.f};
      v8f acc1 = {0.f,0.f,0.f,0.f,0.f,0.f,0.f,0.f};
      #pragma unroll
      for (int kc = 0; kc < KCH; ++kc) {
        if (kc + 2 < KCH) {              // prefetch B fragment kc+2
          const uint4* bp = (const uint4*)(brow + (kc + 2) * 32 + kbB);
          fb[(kc + 2) % 3].q[0] = bp[0];
          fb[(kc + 2) % 3].q[1] = bp[1];
        }
        // one LDS B fragment feeds two back-to-back WMMAs (register M-blocking)
        acc0 = __builtin_amdgcn_wmma_f32_16x16x32_bf16(
            false, a0[kc], false, fb[kc % 3].v, (short)0, acc0, false, false);
        acc1 = __builtin_amdgcn_wmma_f32_16x16x32_bf16(
            false, a1[kc], false, fb[kc % 3].v, (short)0, acc1, false, false);
      }
      #pragma unroll
      for (int i = 0; i < 8; ++i) {      // sim = x.e - 0.5||e||^2 ; keep first max
        float s0 = acc0[i] - h, s1 = acc1[i] - h;
        bool g0_ = s0 > bv0[i];  bi0[i] = g0_ ? code : bi0[i];  bv0[i] = g0_ ? s0 : bv0[i];
        bool g1_ = s1 > bv1[i];  bi1[i] = g1_ ? code : bi1[i];  bv1[i] = g1_ ? s1 : bv1[i];
      }
    }

    // Argmax across the 16 lanes sharing each row (width-16 xor reduce, wave32)
    #pragma unroll
    for (int i = 0; i < 8; ++i) {
      #pragma unroll
      for (int m = 1; m < 16; m <<= 1) {
        float ov; int oi; bool tk;
        ov = __shfl_xor(bv0[i], m, 16); oi = __shfl_xor(bi0[i], m, 16);
        tk = (ov > bv0[i]) || (ov == bv0[i] && oi < bi0[i]);
        bv0[i] = tk ? ov : bv0[i]; bi0[i] = tk ? oi : bi0[i];
        ov = __shfl_xor(bv1[i], m, 16); oi = __shfl_xor(bi1[i], m, 16);
        tk = (ov > bv1[i]) || (ov == bv1[i] && oi < bi1[i]);
        bv1[i] = tk ? ov : bv1[i]; bi1[i] = tk ? oi : bi1[i];
      }
    }
    if (colLn == 0) {                    // lanes 0 & 16: rows i / i+8 (C layout)
      const int half = (lane >= 16) ? 8 : 0;
      #pragma unroll
      for (int i = 0; i < 8; ++i) {
        size_t o0 = (size_t)slice * NN + (row0 + half + i);
        size_t o1 = (size_t)slice * NN + (row0 + 16 + half + i);
        pv[o0] = bv0[i]; pi[o0] = bi0[i];
        pv[o1] = bv1[i]; pi[o1] = bi1[i];
      }
    }
  }
}

// ---------------- Kernel 5: reduce per-slice partials -> final argmax ----------
__global__ __launch_bounds__(256) void argmax_reduce_kernel(
    const float* __restrict__ pv, const int* __restrict__ pi,
    int* __restrict__ idx) {
  int n = blockIdx.x * 256 + threadIdx.x;
  if (n >= NN) return;
  float b = pv[n]; int bi_ = pi[n];
  for (int s = 1; s < SLICES; ++s) {
    float v = pv[(size_t)s * NN + n];
    int   i = pi[(size_t)s * NN + n];
    bool  t = v > b;                     // ascending slices: ties keep lower idx
    b = t ? v : b; bi_ = t ? i : bi_;
  }
  idx[n] = bi_;
}

// ---------------- Kernel 6: per-graph gather/residual/mean/logits --------------
__global__ __launch_bounds__(320) void segment_kernel(
    const float* __restrict__ nf, const float* __restrict__ score,
    const float* __restrict__ cb, const int* __restrict__ idx,
    const int* __restrict__ starts, const float* __restrict__ cls_w,
    const float* __restrict__ cls_b, float* __restrict__ out,
    float* __restrict__ loss_part) {
  __shared__ float cmean[DD];
  __shared__ float lw[10];
  const int g = blockIdx.x;
  const int s0 = starts[g], s1 = starts[g + 1];
  const int d = threadIdx.x;

  float cs = 0.f, ss = 0.f, lsum = 0.f;
  for (int n = s0; n < s1; ++n) {
    const float sc = score[n];
    const int   q  = idx[n];
    if (d < DD) {
      float x = nf[(size_t)n * DD + d];
      float e = cb[(size_t)q * DD + d];  // quantized (fp32 original codebook)
      float res = x + e;                 // node_feat + quantized (ST forward)
      cs += res * sc;
      ss += res * (1.f - sc);
      float df = e - x;
      lsum += df * df;                   // commitment squared error
    }
  }
  const float inv = 1.f / fmaxf((float)(s1 - s0), 1.f);
  if (d < DD) {
    cmean[d] = cs * inv;
    out[(size_t)GG * CC + (size_t)g * DD + d] = cs * inv;                   // c_graph
    out[(size_t)GG * CC + (size_t)GG * DD + (size_t)g * DD + d] = ss * inv; // s_graph
  }
  float l = lsum;
  #pragma unroll
  for (int m = 16; m >= 1; m >>= 1) l += __shfl_xor(l, m, 32);
  if ((threadIdx.x & 31) == 0) lw[threadIdx.x >> 5] = l;
  __syncthreads();
  if (threadIdx.x == 0) {
    float t = 0.f;
    for (int w = 0; w < 10; ++w) t += lw[w];
    loss_part[g] = t;                    // deterministic per-block partial
  }
  if (threadIdx.x < CC) {                // logits: 10 dots of length 300
    float acc = cls_b[threadIdx.x];
    for (int dd = 0; dd < DD; ++dd) acc += cmean[dd] * cls_w[threadIdx.x * DD + dd];
    out[(size_t)g * CC + threadIdx.x] = acc;
  }
}

// ---------------- Kernel 7: deterministic loss reduce --------------------------
__global__ __launch_bounds__(256) void loss_reduce_kernel(
    const float* __restrict__ loss_part, float* __restrict__ out) {
  __shared__ float red[256];
  float s = 0.f;
  for (int g = threadIdx.x; g < GG; g += 256) s += loss_part[g];
  red[threadIdx.x] = s;
  __syncthreads();
  for (int m = 128; m >= 1; m >>= 1) {
    if (threadIdx.x < m) red[threadIdx.x] += red[threadIdx.x + m];
    __syncthreads();
  }
  if (threadIdx.x == 0)
    out[(size_t)GG * CC + 2 * (size_t)GG * DD] =
        COMMIT_WF * red[0] / (float)((size_t)NN * DD);
}

extern "C" void kernel_launch(void* const* d_in, const int* in_sizes, int n_in,
                              void* d_out, int out_size, void* d_ws, size_t ws_size,
                              hipStream_t stream) {
  const float* nf    = (const float*)d_in[0];   // [N, D]
  const float* score = (const float*)d_in[1];   // [N, 1]
  const int*   batch = (const int*)d_in[2];     // [N] sorted graph ids
  const float* cb    = (const float*)d_in[3];   // [K, D]
  const float* clsw  = (const float*)d_in[4];   // [C, D]
  const float* clsb  = (const float*)d_in[5];   // [C]
  float* out = (float*)d_out;

  // Workspace layout (256B-aligned sections), ~16 MB total
  char* ws = (char*)d_ws;
  size_t off = 0;
  unsigned short* cbh = (unsigned short*)(ws + off); off += (size_t)KK * DP * 2;
  float* hn           = (float*)(ws + off);          off += (size_t)KK * 4;
  int* idx            = (int*)(ws + off);            off += (size_t)NN * 4;
  off = (off + 255) & ~(size_t)255;
  float* pv           = (float*)(ws + off);          off += (size_t)SLICES * NN * 4;
  int* pi             = (int*)(ws + off);            off += (size_t)SLICES * NN * 4;
  off = (off + 255) & ~(size_t)255;
  int* starts         = (int*)(ws + off);            off += (size_t)(GG + 1) * 4;
  off = (off + 255) & ~(size_t)255;
  float* loss_part    = (float*)(ws + off);          off += (size_t)GG * 4;
  (void)ws_size; (void)in_sizes; (void)n_in; (void)out_size;

  cb_convert_kernel<<<(KK * DP + 255) / 256, 256, 0, stream>>>(cb, cbh);
  cb_norms_kernel<<<KK / 8, 256, 0, stream>>>(cb, hn);
  seg_starts_kernel<<<(GG + 1 + 255) / 256, 256, 0, stream>>>(batch, starts);

  const size_t vq_lds = ((size_t)KS * DP + (size_t)VQ_WAVES * 2 * 16 * DP) * 2; // 240KB
  dim3 vq_grid(CHUNKS, SLICES);
  vq_argmax_kernel<<<vq_grid, VQ_WAVES * 32, vq_lds, stream>>>(nf, cbh, hn, pv, pi);

  argmax_reduce_kernel<<<(NN + 255) / 256, 256, 0, stream>>>(pv, pi, idx);
  segment_kernel<<<GG, 320, 0, stream>>>(nf, score, cb, idx, starts, clsw, clsb,
                                         out, loss_part);
  loss_reduce_kernel<<<1, 256, 0, stream>>>(loss_part, out);
}
